// Mamba2_70660801954039
// MI455X (gfx1250) — matrix operand order repaired and checked by reading, not compiled, not run on hardware
//
#include <hip/hip_runtime.h>
#include <hip/hip_bf16.h>

// ---------------------------------------------------------------------------
// Mamba2 forward for MI455X (gfx1250, wave32, WMMA bf16 16x16x32, fp32 acc).
//
//   cast u -> bf16 ; cast W_in/W_out -> bf16 TRANSPOSED (N x K) so GEMM
//   B-fragments are contiguous (global_load_b128) instead of strided u16.
//   GEMM1 (WMMA):  zxbcdt = u @ W_in                  (4096 x 2048 x 10304)
//   conv1d+silu  -> x/B/C bf16 buffers
//   softplus(dt)+cumsum(dA) per (chunk, head)
//   SSD intra (WMMA): G=C.B^T, y=(G*M).xdt, S=B^T dec.xdt   per (c,h)
//   chunk scan   : H_{c+1} = dec_c*H_c + S_c  (Hprev stored bf16, p-major)
//   SSD inter (WMMA): y += exp(cum) * (C @ Hprev)
//   gate+RMSNorm -> normed bf16
//   GEMM2 (WMMA):  out = normed @ W_out              (4096 x 4096 x 2048)
// ---------------------------------------------------------------------------

typedef __bf16 bf16_t;
typedef __attribute__((ext_vector_type(16))) __bf16 v16bf;
typedef __attribute__((ext_vector_type(8)))  float  v8f;

#define L_SEQ   4096
#define D_MODEL 2048
#define D_INNER 4096
#define NPROJ   10304      // 2*D_INNER + 2*D_XB + NHEADS
#define D_XB    1024
#define CONV_D  6144       // D_INNER + 2*D_XB
#define NHEADS  64
#define DSTATE  64
#define HDIM    64         // D_INNER / NHEADS
#define NCHUNK  32
#define CHUNKQ  128

// ---------------- WMMA fragment helpers (wave32, ISA 7.12.2 layouts) -------

__device__ inline v8f wmma_bf16(v16bf a, v16bf b, v8f c) {
  return __builtin_amdgcn_wmma_f32_16x16x32_bf16(
      /*neg_a=*/false, a, /*neg_b=*/false, b,
      /*c_mod=*/(short)0, c, /*reuse_a=*/false, /*reuse_b=*/false);
}

// A fragment: 16x32 bf16 tile, row-major source with stride lda.
// lane<16 : elems 0..15 = K 0..15   (row M = lane)
// lane>=16: elems 0..15 = K 16..31  (row M = lane-16)
__device__ inline v16bf load_a_frag(const bf16_t* a, int lda) {
  int lane = threadIdx.x & 31;
  int half = lane >> 4;
  int r    = lane & 15;
  const bf16_t* row = a + (size_t)r * lda;
  v16bf f;
  int k0 = half * 16;
  int k1 = half * 16 + 8;
#pragma unroll
  for (int e = 0; e < 8; ++e) f[e]     = row[k0 + e];
#pragma unroll
  for (int e = 0; e < 8; ++e) f[8 + e] = row[k1 + e];
  return f;
}

// B fragment (32x16, K x N) loaded as the TRANSPOSE of a row-major N x K
// source: B_mat[k][n] = src[n*lds + k]  (contiguous along k -> b128 loads).
__device__ inline v16bf load_bt_frag(const bf16_t* src, int lds) {
  int lane = threadIdx.x & 31;
  int half = lane >> 4;
  int n    = lane & 15;
  const bf16_t* p = src + (size_t)n * lds + half * 16;
  v16bf f;
#pragma unroll
  for (int e = 0; e < 16; ++e) f[e] = p[e];
  return f;
}

// D fragment store: 16x16 f32, VGPR v holds row (v + half*8), col = lane&15.
__device__ inline void store_d_frag(float* d, int ldd, v8f acc) {
  int lane = threadIdx.x & 31;
  int half = lane >> 4;
  int n    = lane & 15;
#pragma unroll
  for (int v = 0; v < 8; ++v) d[(size_t)(v + half * 8) * ldd + n] = acc[v];
}

// ---------------- elementwise / setup kernels ------------------------------

__global__ void cast_bf16_kernel(const float* __restrict__ in,
                                 bf16_t* __restrict__ out, size_t n) {
  size_t i = (size_t)blockIdx.x * blockDim.x + threadIdx.x;
  if (i < n) out[i] = (bf16_t)in[i];
}

// in: R x C row-major f32  ->  out: C x R row-major bf16 (transposed)
__global__ __launch_bounds__(256)
void cast_transpose_bf16_kernel(const float* __restrict__ in,
                                bf16_t* __restrict__ out, int R, int C) {
  __shared__ float tile[32][33];
  int rb = blockIdx.y * 32, cb = blockIdx.x * 32;
  int tx = threadIdx.x & 31, ty = threadIdx.x >> 5;   // 32 x 8
#pragma unroll
  for (int i = ty; i < 32; i += 8) {
    int r = rb + i, c = cb + tx;
    tile[i][tx] = (r < R && c < C) ? in[(size_t)r * C + c] : 0.0f;
  }
  __syncthreads();
#pragma unroll
  for (int i = ty; i < 32; i += 8) {
    int c = cb + i, r = rb + tx;
    if (c < C && r < R) out[(size_t)c * R + r] = (bf16_t)tile[tx][i];
  }
}

// depthwise conv (taps=4) + bias + silu, split into x/B/C bf16 buffers
__global__ void conv_silu_kernel(const float* __restrict__ zx,
                                 const float* __restrict__ cw,
                                 const float* __restrict__ cb,
                                 bf16_t* __restrict__ Xb,
                                 bf16_t* __restrict__ Bb,
                                 bf16_t* __restrict__ Cb) {
  size_t idx = (size_t)blockIdx.x * blockDim.x + threadIdx.x;
  if (idx >= (size_t)L_SEQ * CONV_D) return;
  int l  = (int)(idx / CONV_D);
  int ch = (int)(idx % CONV_D);
  float acc = cb[ch];
#pragma unroll
  for (int k = 0; k < 4; ++k) {
    int ls = l - 3 + k;
    if (ls >= 0) acc += zx[(size_t)ls * NPROJ + D_INNER + ch] * cw[ch * 4 + k];
  }
  float s = acc / (1.0f + __expf(-acc));   // silu
  bf16_t bv = (bf16_t)s;
  if (ch < D_XB)            Xb[(size_t)l * D_XB + ch]          = bv;
  else if (ch < 2 * D_XB)   Bb[(size_t)l * D_XB + (ch - D_XB)] = bv;
  else                      Cb[(size_t)l * D_INNER + (ch - 2 * D_XB)] = bv;
}

// softplus(dt + bias), cumsum of dt*A within each chunk, chunk decay
__global__ void dt_cumsum_kernel(const float* __restrict__ zx,
                                 const float* __restrict__ dt_bias,
                                 const float* __restrict__ A_log,
                                 float* __restrict__ dts,
                                 float* __restrict__ cum,
                                 float* __restrict__ cdec) {
  int idx = blockIdx.x * blockDim.x + threadIdx.x;
  if (idx >= NCHUNK * NHEADS) return;
  int c = idx >> 6, h = idx & 63;
  float Ah   = -__expf(A_log[h]);
  float bias = dt_bias[h];
  float run  = 0.0f;
  for (int q = 0; q < CHUNKQ; ++q) {
    int l = c * CHUNKQ + q;
    float x = zx[(size_t)l * NPROJ + D_INNER + CONV_D + h] + bias;
    float dtv = (x > 20.0f) ? x : log1pf(__expf(x));   // softplus
    dts[(size_t)l * NHEADS + h] = dtv;
    run += dtv * Ah;
    cum[(size_t)l * NHEADS + h] = run;
  }
  cdec[c * NHEADS + h] = __expf(run);
}

// ---------------- SSD intra-chunk (WMMA heavy) -----------------------------
// One block per (chunk, head). 128 threads = 4 waves.
__global__ __launch_bounds__(128)
void ssd_intra_kernel(const bf16_t* __restrict__ Cb,
                      const bf16_t* __restrict__ Bb,
                      const bf16_t* __restrict__ Xb,
                      const float*  __restrict__ dts,
                      const float*  __restrict__ cum,
                      float* __restrict__ S,
                      float* __restrict__ Y) {
  int c  = blockIdx.x;       // chunk
  int h  = blockIdx.y;       // head
  int hb = h >> 2;           // REP = 4: x/B head
  int tid = threadIdx.x;
  int wave = tid >> 5;
  int l0 = c * CHUNKQ;

  __shared__ bf16_t sGm[CHUNKQ * CHUNKQ];   // masked G, bf16 (row-major)
  __shared__ bf16_t sXdtT[HDIM * CHUNKQ];   // (x*dt)^T : p-major, contiguous j
  __shared__ bf16_t sBd[DSTATE * CHUNKQ];   // B^T * decay : n-major
  __shared__ float  sCum[CHUNKQ];
  __shared__ float  sDec[CHUNKQ];

  for (int i = tid; i < CHUNKQ; i += 128)
    sCum[i] = cum[(size_t)(l0 + i) * NHEADS + h];
  __syncthreads();
  float cl = sCum[CHUNKQ - 1];
  for (int i = tid; i < CHUNKQ; i += 128)
    sDec[i] = __expf(cl - sCum[i]);
  __syncthreads();

  for (int idx = tid; idx < CHUNKQ * HDIM; idx += 128) {
    int j = idx >> 6, p = idx & 63;
    float dtv = dts[(size_t)(l0 + j) * NHEADS + h];
    float xv  = (float)Xb[(size_t)(l0 + j) * D_XB + hb * DSTATE + p];
    sXdtT[p * CHUNKQ + j] = (bf16_t)(xv * dtv);
    float bv  = (float)Bb[(size_t)(l0 + j) * D_XB + hb * DSTATE + p];
    sBd[p * CHUNKQ + j] = (bf16_t)(bv * sDec[j]);   // p doubles as n index
  }
  __syncthreads();

  int lane = tid & 31, halfq = lane >> 4, ncol = lane & 15;

  // --- phase 1: G = C (128x64) @ B^T (64x128); mask+decay -> bf16 in LDS ---
  for (int t = wave; t < 64; t += 4) {
    int it = t >> 3, jt = t & 7;
    v8f acc = (v8f)(0.0f);
#pragma unroll
    for (int k = 0; k < DSTATE; k += 32) {
      v16bf a = load_a_frag(Cb + (size_t)(l0 + it * 16) * D_INNER + h * DSTATE + k, D_INNER);
      v16bf b = load_bt_frag(Bb + (size_t)(l0 + jt * 16) * D_XB + hb * DSTATE + k, D_XB);
      acc = wmma_bf16(a, b, acc);
    }
    int j   = jt * 16 + ncol;
    float cj = sCum[j];
#pragma unroll
    for (int v = 0; v < 8; ++v) {
      int i = it * 16 + v + halfq * 8;
      float e = __expf(sCum[i] - cj);          // unconditional -> no branches
      float g = acc[v] * e;
      sGm[i * CHUNKQ + j] = (bf16_t)((j <= i) ? g : 0.0f);   // v_cndmask
    }
  }
  __syncthreads();

  // --- phase 2: y_intra = Gm (128x128) @ xdt (128x64) -> global Y ----------
  for (int t = wave; t < 32; t += 4) {
    int it = t >> 2, pt = t & 3;
    v8f acc = (v8f)(0.0f);
#pragma unroll
    for (int k = 0; k < CHUNKQ; k += 32) {
      v16bf a = load_a_frag(sGm + it * 16 * CHUNKQ + k, CHUNKQ);
      v16bf b = load_bt_frag(sXdtT + (pt * 16) * CHUNKQ + k, CHUNKQ);
      acc = wmma_bf16(a, b, acc);
    }
    store_d_frag(Y + (size_t)(l0 + it * 16) * D_INNER + h * HDIM + pt * 16,
                 D_INNER, acc);
  }

  // --- phase 3: S = (B^T dec) (64x128) @ xdt (128x64) -> global state ------
  for (int t = wave; t < 16; t += 4) {
    int nt = t >> 2, pt = t & 3;
    v8f acc = (v8f)(0.0f);
#pragma unroll
    for (int k = 0; k < CHUNKQ; k += 32) {
      v16bf a = load_a_frag(sBd + nt * 16 * CHUNKQ + k, CHUNKQ);
      v16bf b = load_bt_frag(sXdtT + (pt * 16) * CHUNKQ + k, CHUNKQ);
      acc = wmma_bf16(a, b, acc);
    }
    store_d_frag(S + (((size_t)c * NHEADS + h) * DSTATE + nt * 16) * HDIM + pt * 16,
                 HDIM, acc);
  }
}

// ---------------- inter-chunk recurrence (sequential over 32 chunks) -------
// Hprev written TRANSPOSED: layout (c, h, p, n) so inter B-frags are contiguous.
__global__ void scan_kernel(const float* __restrict__ S,
                            const float* __restrict__ cdec,
                            bf16_t* __restrict__ HprevT) {
  int idx = blockIdx.x * blockDim.x + threadIdx.x;   // h*4096 + n*64 + p
  if (idx >= NHEADS * DSTATE * HDIM) return;
  int h = idx >> 12;
  int n = (idx >> 6) & 63;
  int p = idx & 63;
  float H = 0.0f;
  for (int c = 0; c < NCHUNK; ++c) {
    HprevT[(((size_t)c * NHEADS + h) * HDIM + p) * DSTATE + n] = (bf16_t)H;
    H = cdec[c * NHEADS + h] * H + S[(size_t)c * NHEADS * DSTATE * HDIM + idx];
  }
}

// ---------------- SSD inter-chunk correction: y += exp(cum)*(C @ Hprev) ----
__global__ __launch_bounds__(128)
void ssd_inter_kernel(const bf16_t* __restrict__ Cb,
                      const bf16_t* __restrict__ HprevT,
                      const float*  __restrict__ cum,
                      float* __restrict__ Y) {
  int c = blockIdx.x, h = blockIdx.y;
  int tid = threadIdx.x, wave = tid >> 5;
  int l0 = c * CHUNKQ;
  __shared__ float sCum[CHUNKQ];
  for (int i = tid; i < CHUNKQ; i += 128)
    sCum[i] = cum[(size_t)(l0 + i) * NHEADS + h];
  __syncthreads();

  const bf16_t* Ht = HprevT + ((size_t)c * NHEADS + h) * HDIM * DSTATE;
  int lane = tid & 31, halfq = lane >> 4, ncol = lane & 15;

  for (int t = wave; t < 32; t += 4) {
    int it = t >> 2, pt = t & 3;
    v8f acc = (v8f)(0.0f);
#pragma unroll
    for (int k = 0; k < DSTATE; k += 32) {
      v16bf a = load_a_frag(Cb + (size_t)(l0 + it * 16) * D_INNER + h * DSTATE + k, D_INNER);
      v16bf b = load_bt_frag(Ht + (size_t)(pt * 16) * DSTATE + k, DSTATE);
      acc = wmma_bf16(a, b, acc);
    }
    float* yp = Y + (size_t)(l0 + it * 16) * D_INNER + h * HDIM + pt * 16;
#pragma unroll
    for (int v = 0; v < 8; ++v) {
      int i = it * 16 + v + halfq * 8;
      yp[(size_t)(v + halfq * 8) * D_INNER + ncol] += __expf(sCum[i]) * acc[v];
    }
  }
}

// ---------------- gating (silu(z)) + per-head RMSNorm + weight -> bf16 -----
// Two-pass (Y is L2-resident) to avoid a 64-deep private array spilling.
__global__ void gate_norm_kernel(const float* __restrict__ Y,
                                 const bf16_t* __restrict__ Xb,
                                 const float* __restrict__ zx,
                                 const float* __restrict__ Dp,
                                 const float* __restrict__ nw,
                                 bf16_t* __restrict__ normed) {
  int idx = blockIdx.x * blockDim.x + threadIdx.x;   // l*64 + h
  if (idx >= L_SEQ * NHEADS) return;
  int l = idx >> 6, h = idx & 63;
  int hb = h >> 2;
  float Dv = Dp[h];
  const float*  yp = Y  + (size_t)l * D_INNER + h * HDIM;
  const bf16_t* xp = Xb + (size_t)l * D_XB   + hb * DSTATE;
  const float*  zp = zx + (size_t)l * NPROJ  + h * HDIM;   // z = first D_INNER
  float ss = 0.0f;
#pragma unroll 4
  for (int p = 0; p < HDIM; ++p) {
    float yv = yp[p] + (float)xp[p] * Dv;
    float zv = zp[p];
    float v  = yv * (zv / (1.0f + __expf(-zv)));
    ss += v * v;
  }
  float r = rsqrtf(ss * (1.0f / HDIM) + 1e-5f);
  bf16_t* op = normed + (size_t)l * D_INNER + h * HDIM;
#pragma unroll 4
  for (int p = 0; p < HDIM; ++p) {
    float yv = yp[p] + (float)xp[p] * Dv;
    float zv = zp[p];
    float v  = yv * (zv / (1.0f + __expf(-zv)));
    op[p] = (bf16_t)(v * r * nw[h * HDIM + p]);
  }
}

// ---------------- dense GEMM (WMMA): C[f32] = A[bf16] @ Bt[bf16]^T ---------
// A: M x K row-major.  Bt: N x K row-major (pre-transposed weights).
// grid = (N/64, M/128), block = 128 (4 waves); wave -> 32 rows x 64 cols.
// global_prefetch of the next K-tile hides L2 latency on the weight stream.
__global__ __launch_bounds__(128)
void gemm_bf16_tn_kernel(const bf16_t* __restrict__ A,
                         const bf16_t* __restrict__ Bt,
                         float* __restrict__ C, int M, int N, int K) {
  int wave = threadIdx.x >> 5;
  int lane = threadIdx.x & 31;
  int row0 = blockIdx.y * 128 + wave * 32;
  int col0 = blockIdx.x * 64;
  v8f acc[2][4];
#pragma unroll
  for (int r = 0; r < 2; ++r)
#pragma unroll
    for (int t = 0; t < 4; ++t) acc[r][t] = (v8f)(0.0f);

#pragma unroll 2
  for (int k = 0; k < K; k += 32) {
    // prefetch next K-tile (speculative; lanes cover the 32 rows of A strip
    // and the 64 rows of the Bt strips one iteration ahead)
    if (k + 64 < K) {
      __builtin_prefetch(A  + (size_t)(row0 + lane) * K + k + 64, 0, 3);
      __builtin_prefetch(Bt + (size_t)(col0 + lane) * K + k + 64, 0, 3);
      __builtin_prefetch(Bt + (size_t)(col0 + 32 + lane) * K + k + 64, 0, 3);
    }
    v16bf a0 = load_a_frag(A + (size_t)row0 * K + k, K);
    v16bf a1 = load_a_frag(A + (size_t)(row0 + 16) * K + k, K);
#pragma unroll
    for (int t = 0; t < 4; ++t) {
      v16bf b = load_bt_frag(Bt + (size_t)(col0 + t * 16) * K + k, K);
      acc[0][t] = wmma_bf16(a0, b, acc[0][t]);
      acc[1][t] = wmma_bf16(a1, b, acc[1][t]);
    }
  }
#pragma unroll
  for (int r = 0; r < 2; ++r)
#pragma unroll
    for (int t = 0; t < 4; ++t)
      store_d_frag(C + (size_t)(row0 + r * 16) * N + col0 + t * 16, N, acc[r][t]);
}

// ---------------------------------------------------------------------------

extern "C" void kernel_launch(void* const* d_in, const int* in_sizes, int n_in,
                              void* d_out, int out_size, void* d_ws, size_t ws_size,
                              hipStream_t stream) {
  const float* u       = (const float*)d_in[0];
  const float* W_in    = (const float*)d_in[1];
  const float* conv_w  = (const float*)d_in[2];
  const float* conv_b  = (const float*)d_in[3];
  const float* dt_bias = (const float*)d_in[4];
  const float* A_log   = (const float*)d_in[5];
  const float* Dp      = (const float*)d_in[6];
  const float* nw      = (const float*)d_in[7];
  const float* W_out   = (const float*)d_in[8];
  float* out = (float*)d_out;

  // ---- workspace carve (~448 MB) ----
  char* wp = (char*)d_ws;
  auto carve = [&](size_t bytes) -> void* {
    void* r = (void*)wp;
    wp += (bytes + 255) & ~(size_t)255;
    return r;
  };
  bf16_t* u_bf    = (bf16_t*)carve((size_t)L_SEQ * D_MODEL * 2);
  bf16_t* WinT_bf = (bf16_t*)carve((size_t)D_MODEL * NPROJ * 2);   // NPROJ x D_MODEL
  bf16_t* WoutT_bf= (bf16_t*)carve((size_t)D_INNER * D_MODEL * 2); // D_MODEL x D_INNER
  float*  zx      = (float*) carve((size_t)L_SEQ * NPROJ * 4);
  bf16_t* Xb      = (bf16_t*)carve((size_t)L_SEQ * D_XB * 2);
  bf16_t* Bb      = (bf16_t*)carve((size_t)L_SEQ * D_XB * 2);
  bf16_t* Cb      = (bf16_t*)carve((size_t)L_SEQ * D_INNER * 2);
  float*  dts     = (float*) carve((size_t)L_SEQ * NHEADS * 4);
  float*  cum     = (float*) carve((size_t)L_SEQ * NHEADS * 4);
  float*  cdec    = (float*) carve((size_t)NCHUNK * NHEADS * 4);
  float*  Sst     = (float*) carve((size_t)NCHUNK * NHEADS * DSTATE * HDIM * 4);
  bf16_t* HprevT  = (bf16_t*)carve((size_t)NCHUNK * NHEADS * DSTATE * HDIM * 2);
  float*  Ybuf    = (float*) carve((size_t)L_SEQ * D_INNER * 4);
  bf16_t* normed  = (bf16_t*)carve((size_t)L_SEQ * D_INNER * 2);

  // ---- casts (weights transposed for contiguous WMMA B-fragments) ----
  {
    size_t n = (size_t)L_SEQ * D_MODEL;
    cast_bf16_kernel<<<(n + 255) / 256, 256, 0, stream>>>(u, u_bf, n);
  }
  cast_transpose_bf16_kernel<<<dim3((NPROJ + 31) / 32, (D_MODEL + 31) / 32),
                               256, 0, stream>>>(W_in, WinT_bf, D_MODEL, NPROJ);
  cast_transpose_bf16_kernel<<<dim3((D_MODEL + 31) / 32, (D_INNER + 31) / 32),
                               256, 0, stream>>>(W_out, WoutT_bf, D_INNER, D_MODEL);

  // ---- GEMM1: zxbcdt = u @ W_in ----
  gemm_bf16_tn_kernel<<<dim3(NPROJ / 64, L_SEQ / 128), 128, 0, stream>>>(
      u_bf, WinT_bf, zx, L_SEQ, NPROJ, D_MODEL);

  // ---- conv + silu ----
  {
    size_t n = (size_t)L_SEQ * CONV_D;
    conv_silu_kernel<<<(n + 255) / 256, 256, 0, stream>>>(zx, conv_w, conv_b,
                                                          Xb, Bb, Cb);
  }

  // ---- dt softplus + cumsum ----
  dt_cumsum_kernel<<<(NCHUNK * NHEADS + 255) / 256, 256, 0, stream>>>(
      zx, dt_bias, A_log, dts, cum, cdec);

  // ---- SSD intra-chunk (WMMA) ----
  ssd_intra_kernel<<<dim3(NCHUNK, NHEADS), 128, 0, stream>>>(
      Cb, Bb, Xb, dts, cum, Sst, Ybuf);

  // ---- inter-chunk scan ----
  {
    int n = NHEADS * DSTATE * HDIM;
    scan_kernel<<<(n + 255) / 256, 256, 0, stream>>>(Sst, cdec, HprevT);
  }

  // ---- SSD inter correction (WMMA) ----
  ssd_inter_kernel<<<dim3(NCHUNK, NHEADS), 128, 0, stream>>>(Cb, HprevT, cum, Ybuf);

  // ---- gate + RMSNorm ----
  {
    int n = L_SEQ * NHEADS;
    gate_norm_kernel<<<(n + 255) / 256, 256, 0, stream>>>(Ybuf, Xb, zx, Dp, nw,
                                                          normed);
  }

  // ---- GEMM2: out = normed @ W_out ----
  gemm_bf16_tn_kernel<<<dim3(D_MODEL / 64, L_SEQ / 128), 128, 0, stream>>>(
      normed, WoutT_bf, out, L_SEQ, D_MODEL, D_INNER);
}
